// CEGCN_43937515438274
// MI455X (gfx1250) — matrix operand order repaired and checked by reading, not compiled, not run on hardware
//
#include <hip/hip_runtime.h>
#include <hip/hip_bf16.h>
#include <stdint.h>

// ---------------------------------------------------------------------------
// CEGCN pipeline for MI455X (gfx1250, wave32, WMMA).
// HBM floor: streaming Q1/Q2 (512 MB each, read once) ~45us at 23.3 TB/s.
// All GEMMs: v_wmma_f32_16x16x32_f16 (f16 operands, f32 accumulate).
// B operands live in WMMA-fragment order (packed by producers) -> fragment
// load = 2x global_load_b128; A fragments = 2x ds_load_b128 from LDS.
// f16 A-tiles staged with GLOBAL_LOAD_ASYNC_TO_LDS (ASYNCcnt).
// qtx (the Q stream) is K-split 4x for memory-latency concurrency and
// accumulates via global_atomic_add_f32; prefetch runs 4 slabs ahead.
// ---------------------------------------------------------------------------

typedef __attribute__((ext_vector_type(16))) _Float16 v16h;
typedef __attribute__((ext_vector_type(8)))  float    v8f;

#define N_SP        2048
#define HW_PIX      65536
#define C_IN        128
#define LEAKY_SLOPE 0.01f
#define BN_EPS      1e-5f
#define KSTEP       64
#define QTX_SPLIT   4
#define PF_DIST     (4 * KSTEP)

#define USE_ASYNC_COPY 1   // gfx1250 global_load_async_to_lds_b32 staging

__device__ __forceinline__ v8f wmma_f16(v16h a, v16h b, v8f c) {
  return __builtin_amdgcn_wmma_f32_16x16x32_f16(false, a, false, b,
                                                (short)0, c, false, false);
}

// Async global->LDS 4-byte copy (per lane). LDS address = low 32 bits of the
// generic shared pointer.
__device__ __forceinline__ void async_copy_b32(const void* gptr, void* lptr) {
#if USE_ASYNC_COPY
  uint32_t lds_off = (uint32_t)(uintptr_t)lptr;
  asm volatile("global_load_async_to_lds_b32 %0, %1, off"
               :: "v"(lds_off), "v"(gptr) : "memory");
#else
  *(uint32_t*)lptr = *(const uint32_t*)gptr;
#endif
}
__device__ __forceinline__ void async_copy_wait() {
#if USE_ASYNC_COPY
  asm volatile("s_wait_asynccnt 0" ::: "memory");
#endif
}

// --- B fragment packing ----------------------------------------------------
// Packed layout: tile(kt=k>>5, ntile=n>>4) of 512 halves; within a tile:
// lane = 16*((k>>4)&1) + (n&15), element e = k&15.
__device__ __forceinline__ size_t bpack_idx(int k, int n, int N) {
  return ((size_t)(k >> 5) * (N >> 4) + (n >> 4)) * 512 +
         (size_t)((((k >> 4) & 1) << 4) + (n & 15)) * 16 + (k & 15);
}

// One v16h fragment = 32 contiguous, 32B-aligned bytes per lane.
__device__ __forceinline__ v16h load_b_packed(const _Float16* Bp, int N, int k0,
                                              int n0, int lane) {
  const _Float16* p = Bp + ((size_t)(k0 >> 5) * (N >> 4) + (n0 >> 4)) * 512 +
                      (size_t)lane * 16;
  return *(const v16h*)p;   // 2x global_load_b128
}

// A fragment (16x32 f16) from row-major f16 LDS tile: two 16B runs.
__device__ __forceinline__ v16h load_a_f16(const _Float16* S, int lda, int m0,
                                           int k0, int lane) {
  const _Float16* row = S + (size_t)(m0 + (lane & 15)) * lda + k0 + ((lane >> 4) << 4);
  union { v16h h; uint4 q[2]; } r;
  r.q[0] = *(const uint4*)(row);        // halves 0..7
  r.q[1] = *(const uint4*)(row + 8);    // halves 8..15
  return r.h;
}

// ---------------------------------------------------------------------------
// Partial H[N_SP, C_IN] += Q^T @ X over a K-slice; column sums of Q
// accumulated in f32 during staging (ds_add_f32 + one global atomic/col).
// Grid: (m-tiles = 128, k-slices = QTX_SPLIT). H and csum must be zeroed.
// ---------------------------------------------------------------------------
__global__ void __launch_bounds__(256) qtx_kernel(const float* __restrict__ Q,
                                                  const _Float16* __restrict__ Xp,
                                                  float* __restrict__ Hacc,
                                                  float* __restrict__ csum) {
  __shared__ _Float16 sQ[16][KSTEP + 8];   // [m][k] row-major, padded
  __shared__ float scol[16];
  const int tid = threadIdx.x, lane = tid & 31, wave = tid >> 5;
  const int m0 = blockIdx.x * 16;
  const int n0 = wave * 16;                // 8 waves cover C=128
  const int kbeg = blockIdx.y * (HW_PIX / QTX_SPLIT);
  const int kend = kbeg + (HW_PIX / QTX_SPLIT);
  const int mm = (tid & 7) * 2;            // this thread's two Q columns
  v8f acc = {};
  float cs0 = 0.0f, cs1 = 0.0f;

  for (int k0 = kbeg; k0 < kend; k0 += KSTEP) {
    // stage Q[k0..k0+63, m0..m0+15] -> sQ[m][k] (transpose + f16 cvt),
    // accumulating f32 column sums on the way through.
#pragma unroll
    for (int i = 0; i < 2; ++i) {
      int idx = tid + i * 256;             // 0..511, 2 floats each
      int kk = idx >> 3;                   // (idx&7) == (tid&7) for both i
      float2 q = *(const float2*)&Q[(size_t)(k0 + kk) * N_SP + m0 + mm];
      cs0 += q.x; cs1 += q.y;
      sQ[mm][kk]     = (_Float16)q.x;
      sQ[mm + 1][kk] = (_Float16)q.y;
    }
    if (tid < KSTEP && k0 + PF_DIST < kend)   // 4 slabs ahead; 64B line == tile row
      __builtin_prefetch(&Q[(size_t)(k0 + PF_DIST + tid) * N_SP + m0], 0, 1);
    __syncthreads();
#pragma unroll
    for (int kk = 0; kk < KSTEP; kk += 32) {
      v16h a = load_a_f16(&sQ[0][0], KSTEP + 8, 0, kk, lane);
      v16h b = load_b_packed(Xp, C_IN, k0 + kk, n0, lane);
      acc = wmma_f16(a, b, acc);
    }
    __syncthreads();
  }
  // column-sum reduction: LDS atomics, then one global atomic per column
  if (tid < 16) scol[tid] = 0.0f;
  __syncthreads();
  atomicAdd(&scol[mm],     cs0);
  atomicAdd(&scol[mm + 1], cs1);
  __syncthreads();
  if (tid < 16) atomicAdd(&csum[m0 + tid], scol[tid]);
  // partial H accumulation
  const int n = n0 + (lane & 15);
  const int h = lane >> 4;
#pragma unroll
  for (int r = 0; r < 8; ++r)
    atomicAdd(&Hacc[(size_t)(m0 + h * 8 + r) * C_IN + n], acc[r]);
}

// H[m, c] /= csum[m]   (finishes the superpixel mean)
__global__ void hnorm_kernel(float* __restrict__ H, const float* __restrict__ csum) {
  int i = blockIdx.x * 256 + threadIdx.x;       // over N_SP * C_IN
  H[i] = H[i] / csum[i >> 7];
}

__global__ void zero_kernel(float* __restrict__ p, int n) {
  int i = blockIdx.x * 256 + threadIdx.x;
  if (i < n) p[i] = 0.0f;
}

// ---------------------------------------------------------------------------
// Generic C[M,N] = act(A[M,K] @ B[K,N] + bias).
// A16: A is f16 row-major (async-copied to LDS); else f32 (cvt in staging,
// prefetch 4 slabs ahead). B always packed f16. OUT16 writes packed f16.
// Block = 32(M) x 64(N), 8 waves (2x4); inactive N-tiles masked wave-
// uniformly so EXEC stays full for WMMA. M%32==0, K%64==0, N%16==0.
// ---------------------------------------------------------------------------
template <bool A16, bool BIAS, bool RELU, bool OUT16>
__global__ void __launch_bounds__(256) gemm_kernel(const void* __restrict__ Av,
                                                   const _Float16* __restrict__ B,
                                                   const float* __restrict__ bias,
                                                   void* __restrict__ Cv,
                                                   int M, int N, int K) {
  __shared__ _Float16 sA[32][KSTEP + 8];   // 32 x 72 halves
  const int tid = threadIdx.x, lane = tid & 31, wave = tid >> 5;
  const int mb = blockIdx.x * 32;
  const int nb = blockIdx.y * 64;
  const int mt = (wave >> 2) << 4;
  const int nt = nb + ((wave & 3) << 4);
  const bool active = (nt < N);
  v8f acc = {};
  for (int k0 = 0; k0 < K; k0 += KSTEP) {
    if (A16) {
      const _Float16* A = (const _Float16*)Av;
#pragma unroll
      for (int i = 0; i < 4; ++i) {
        int idx = tid + i * 256;           // 1024 dwords = 32 x 64 halves
        int r = idx >> 5, c = (idx & 31) * 2;
        async_copy_b32(&A[(size_t)(mb + r) * K + k0 + c], &sA[r][c]);
      }
      async_copy_wait();
    } else {
      const float* A = (const float*)Av;
#pragma unroll
      for (int i = 0; i < 4; ++i) {
        int idx = tid + i * 256;           // 1024 float-pairs
        int r = idx >> 5, c = (idx & 31) * 2;
        float2 q = *(const float2*)&A[(size_t)(mb + r) * K + k0 + c];
        union { _Float16 h[2]; uint32_t u; } p;
        p.h[0] = (_Float16)q.x; p.h[1] = (_Float16)q.y;
        *(uint32_t*)&sA[r][c] = p.u;
      }
      // 4 slabs ahead, 4 cachelines per 64-col row: 128 threads cover it
      if (tid < 128 && k0 + PF_DIST < K)
        __builtin_prefetch(&A[(size_t)(mb + (tid >> 2)) * K + k0 + PF_DIST +
                              (tid & 3) * 16], 0, 1);
    }
    __syncthreads();
    if (active) {
#pragma unroll
      for (int kk = 0; kk < KSTEP; kk += 32) {
        v16h a = load_a_f16(&sA[0][0], KSTEP + 8, mt, kk, lane);
        v16h b = load_b_packed(B, N, k0 + kk, nt, lane);
        acc = wmma_f16(a, b, acc);
      }
    }
    __syncthreads();
  }
  if (active) {
    int n = nt + (lane & 15);
    float bv = BIAS ? bias[n] : 0.0f;
    int h = lane >> 4;
#pragma unroll
    for (int r = 0; r < 8; ++r) {
      float v = acc[r] + bv;
      if (RELU) v = (v > 0.0f) ? v : v * LEAKY_SLOPE;
      int krow = mb + mt + h * 8 + r;
      if (OUT16) ((_Float16*)Cv)[bpack_idx(krow, n, N)] = (_Float16)v;
      else       ((float*)Cv)[(size_t)krow * N + n]     = v;
    }
  }
}

// One-shot f32 -> packed-f16 conversion for B operands (weights, X).
// N is a power of two (128/512/1024/32), so / and % become shifts.
__global__ void cvt_pack_b_kernel(const float* __restrict__ in,
                                  _Float16* __restrict__ out, int K, int N) {
  int i = blockIdx.x * 256 + threadIdx.x;
  if (i >= K * N) return;
  int k = i / N, n = i % N;
  out[bpack_idx(k, n, N)] = (_Float16)in[i];
}

// d[i] = rsqrt(rowsum(A)[i] + 1)
__global__ void adj_rowsum_kernel(const float* __restrict__ A, float* __restrict__ dinv) {
  __shared__ float red[256];
  const int row = blockIdx.x, tid = threadIdx.x;
  float s = 0.0f;
  for (int j = tid; j < N_SP; j += 256) s += A[(size_t)row * N_SP + j];
  red[tid] = s;
  __syncthreads();
  for (int st = 128; st > 0; st >>= 1) {
    if (tid < st) red[tid] += red[tid + st];
    __syncthreads();
  }
  if (tid == 0) dinv[row] = rsqrtf(red[0] + 1.0f);
}

// A_hat[i,j] = (A[i,j] + (i==j)) * d[i] * d[j]  -> f16 row-major (WMMA A op)
__global__ void norm_adj_kernel(const float* __restrict__ A,
                                const float* __restrict__ dinv,
                                _Float16* __restrict__ Ah) {
  const int i = blockIdx.x;
  const float di = dinv[i];
  for (int j = threadIdx.x; j < N_SP; j += blockDim.x) {
    float v = A[(size_t)i * N_SP + j] + (i == j ? 1.0f : 0.0f);
    Ah[(size_t)i * N_SP + j] = (_Float16)(v * di * dinv[j]);
  }
}

// Column standardization: out = (in*g - mean) / sqrt(var + eps).
// g==null, eps=BN_EPS -> BatchNorm1d; g!=null, eps=0 -> StandardScaler(X*g).
// OutT = _Float16 when the consumer is a WMMA A-operand. In-place safe.
template <typename OutT>
__global__ void colnorm_kernel(const float* __restrict__ in, int ldi,
                               const float* __restrict__ g,
                               OutT* __restrict__ out, int ldo,
                               int rows, int cols, float eps) {
  const int c = blockIdx.x * blockDim.x + threadIdx.x;
  if (c >= cols) return;
  float s = 0.0f, s2 = 0.0f;
  for (int r = 0; r < rows; ++r) {
    float v = in[(size_t)r * ldi + c];
    if (g) v *= g[r];
    s += v; s2 += v * v;
  }
  const float inv_n = 1.0f / (float)rows;
  const float mu  = s * inv_n;
  const float var = fmaxf(s2 * inv_n - mu * mu, 0.0f);
  const float isd = rsqrtf(var + eps);
  for (int r = 0; r < rows; ++r) {
    float v = in[(size_t)r * ldi + c];
    if (g) v *= g[r];
    out[(size_t)r * ldo + c] = (OutT)((v - mu) * isd);
  }
}

// g[i] = exp(10 * (1 - cos(H1_i, H2_i))) - 1
__global__ void crossgate_kernel(const float* __restrict__ H1,
                                 const float* __restrict__ H2,
                                 float* __restrict__ g, int F) {
  __shared__ float sa[256], sb[256], sd[256];
  const int row = blockIdx.x, tid = threadIdx.x;
  float a = 0.0f, b = 0.0f, d = 0.0f;
  for (int j = tid; j < F; j += 256) {
    float x = H1[(size_t)row * F + j];
    float y = H2[(size_t)row * F + j];
    a += x * x; b += y * y; d += x * y;
  }
  sa[tid] = a; sb[tid] = b; sd[tid] = d;
  __syncthreads();
  for (int st = 128; st > 0; st >>= 1) {
    if (tid < st) { sa[tid] += sa[tid + st]; sb[tid] += sb[tid + st]; sd[tid] += sd[tid + st]; }
    __syncthreads();
  }
  if (tid == 0) {
    float mod  = sqrtf(sa[0]) * sqrtf(sb[0]);
    float cosd = 1.0f - sd[0] / mod;
    g[row] = expf(10.0f * cosd) - 1.0f;
  }
}

// Y = softmax(R @ W_cls + b_cls), R:[HW,32] f32, W_cls:[32,2]
__global__ void cls_softmax_kernel(const float* __restrict__ R,
                                   const float* __restrict__ Wc,
                                   const float* __restrict__ bc,
                                   float* __restrict__ Y) {
  const int i = blockIdx.x * blockDim.x + threadIdx.x;
  if (i >= HW_PIX) return;
  float l0 = bc[0], l1 = bc[1];
#pragma unroll
  for (int j = 0; j < 32; ++j) {
    float v = R[(size_t)i * 32 + j];
    l0 += v * Wc[j * 2 + 0];
    l1 += v * Wc[j * 2 + 1];
  }
  float mx = fmaxf(l0, l1);
  float e0 = expf(l0 - mx), e1 = expf(l1 - mx);
  float inv = 1.0f / (e0 + e1);
  Y[(size_t)i * 2 + 0] = e0 * inv;
  Y[(size_t)i * 2 + 1] = e1 * inv;
}

// ---------------------------------------------------------------------------
extern "C" void kernel_launch(void* const* d_in, const int* in_sizes, int n_in,
                              void* d_out, int out_size, void* d_ws, size_t ws_size,
                              hipStream_t stream) {
  (void)in_sizes; (void)n_in; (void)out_size; (void)ws_size;
  const float* x1    = (const float*)d_in[0];
  const float* x2    = (const float*)d_in[1];
  const float* Q1    = (const float*)d_in[2];
  const float* Q2    = (const float*)d_in[3];
  const float* A1    = (const float*)d_in[4];
  const float* A2    = (const float*)d_in[5];
  const float* W1_0  = (const float*)d_in[6];
  const float* b1_0  = (const float*)d_in[7];
  const float* W1_1  = (const float*)d_in[8];
  const float* b1_1  = (const float*)d_in[9];
  const float* W2_0  = (const float*)d_in[10];
  const float* b2_0  = (const float*)d_in[11];
  const float* W2_1  = (const float*)d_in[12];
  const float* b2_1  = (const float*)d_in[13];
  const float* W_out = (const float*)d_in[14];
  const float* b_out = (const float*)d_in[15];
  const float* W_cls = (const float*)d_in[16];
  const float* b_cls = (const float*)d_in[17];
  float* Y = (float*)d_out;

  // Workspace layout (~83 MB; all intermediates L2-resident).
  char* ws = (char*)d_ws;
  size_t off = 0;
  auto alloc = [&](size_t bytes) {
    void* p = ws + off; off += (bytes + 255) & ~(size_t)255; return p;
  };
  _Float16* Xp1   = (_Float16*)alloc((size_t)HW_PIX * C_IN * 2);   // packed B
  _Float16* Xp2   = (_Float16*)alloc((size_t)HW_PIX * C_IN * 2);   // packed B
  _Float16* A1h   = (_Float16*)alloc((size_t)N_SP * N_SP * 2);     // row-major A
  _Float16* A2h   = (_Float16*)alloc((size_t)N_SP * N_SP * 2);
  float*    H1    = (float*)   alloc((size_t)N_SP * 1024 * 4);
  float*    H2    = (float*)   alloc((size_t)N_SP * 1024 * 4);
  _Float16* T1h   = (_Float16*)alloc((size_t)N_SP * 1024 * 2);     // row-major A
  _Float16* T2h   = (_Float16*)alloc((size_t)N_SP * 1024 * 2);     // packed B
  _Float16* Hch   = (_Float16*)alloc((size_t)N_SP * N_SP * 2);     // row-major A
  _Float16* W10h  = (_Float16*)alloc((size_t)C_IN * 512 * 2);      // packed B
  _Float16* W11h  = (_Float16*)alloc((size_t)512 * 1024 * 2);
  _Float16* W20h  = (_Float16*)alloc((size_t)C_IN * 512 * 2);
  _Float16* W21h  = (_Float16*)alloc((size_t)512 * 1024 * 2);
  _Float16* Wouth = (_Float16*)alloc((size_t)N_SP * 32 * 2);
  _Float16* Touth = (_Float16*)alloc((size_t)N_SP * 32 * 2);       // packed B
  _Float16* Houth = (_Float16*)alloc((size_t)N_SP * 32 * 2);       // packed B
  float*    d1    = (float*)   alloc(N_SP * 4);
  float*    d2    = (float*)   alloc(N_SP * 4);
  float*    g     = (float*)   alloc(N_SP * 4);
  float*    cs1   = (float*)   alloc(N_SP * 4);
  float*    cs2   = (float*)   alloc(N_SP * 4);
  float*    R     = (float*)Xp1;   // X dead after qtx; reuse for [HW,32] f32

  // 0. One-shot packed-f16 conversions + accumulator zeroing
  cvt_pack_b_kernel<<<(HW_PIX * C_IN) / 256, 256, 0, stream>>>(x1, Xp1, HW_PIX, C_IN);
  cvt_pack_b_kernel<<<(HW_PIX * C_IN) / 256, 256, 0, stream>>>(x2, Xp2, HW_PIX, C_IN);
  cvt_pack_b_kernel<<<(C_IN * 512) / 256, 256, 0, stream>>>(W1_0, W10h, C_IN, 512);
  cvt_pack_b_kernel<<<(512 * 1024) / 256, 256, 0, stream>>>(W1_1, W11h, 512, 1024);
  cvt_pack_b_kernel<<<(C_IN * 512) / 256, 256, 0, stream>>>(W2_0, W20h, C_IN, 512);
  cvt_pack_b_kernel<<<(512 * 1024) / 256, 256, 0, stream>>>(W2_1, W21h, 512, 1024);
  cvt_pack_b_kernel<<<(N_SP * 32) / 256, 256, 0, stream>>>(W_out, Wouth, N_SP, 32);
  zero_kernel<<<(N_SP * C_IN) / 256, 256, 0, stream>>>(H1, N_SP * C_IN);
  zero_kernel<<<(N_SP * C_IN) / 256, 256, 0, stream>>>(H2, N_SP * C_IN);
  zero_kernel<<<N_SP / 256, 256, 0, stream>>>(cs1, N_SP);
  zero_kernel<<<N_SP / 256, 256, 0, stream>>>(cs2, N_SP);

  // 1. Superpixel mean features (streams Q once each, K-split 4x)
  qtx_kernel<<<dim3(N_SP / 16, QTX_SPLIT), 256, 0, stream>>>(Q1, Xp1, H1, cs1);
  qtx_kernel<<<dim3(N_SP / 16, QTX_SPLIT), 256, 0, stream>>>(Q2, Xp2, H2, cs2);
  hnorm_kernel<<<(N_SP * C_IN) / 256, 256, 0, stream>>>(H1, cs1);
  hnorm_kernel<<<(N_SP * C_IN) / 256, 256, 0, stream>>>(H2, cs2);

  // 2. Normalized adjacencies (f16)
  adj_rowsum_kernel<<<N_SP, 256, 0, stream>>>(A1, d1);
  adj_rowsum_kernel<<<N_SP, 256, 0, stream>>>(A2, d2);
  norm_adj_kernel<<<N_SP, 256, 0, stream>>>(A1, d1, A1h);
  norm_adj_kernel<<<N_SP, 256, 0, stream>>>(A2, d2, A2h);

  // 3. GCN layer 1 (128 -> 512)
  colnorm_kernel<_Float16><<<1, 128, 0, stream>>>(H1, C_IN, nullptr, T1h, C_IN, N_SP, C_IN, BN_EPS);
  gemm_kernel<true, true,  false, true ><<<dim3(N_SP / 32, 512 / 64), 256, 0, stream>>>(T1h, W10h, b1_0, T2h, N_SP, 512, C_IN);
  gemm_kernel<true, false, true,  false><<<dim3(N_SP / 32, 512 / 64), 256, 0, stream>>>(A1h, T2h, nullptr, H1, N_SP, 512, N_SP);
  colnorm_kernel<_Float16><<<1, 128, 0, stream>>>(H2, C_IN, nullptr, T1h, C_IN, N_SP, C_IN, BN_EPS);
  gemm_kernel<true, true,  false, true ><<<dim3(N_SP / 32, 512 / 64), 256, 0, stream>>>(T1h, W20h, b2_0, T2h, N_SP, 512, C_IN);
  gemm_kernel<true, false, true,  false><<<dim3(N_SP / 32, 512 / 64), 256, 0, stream>>>(A2h, T2h, nullptr, H2, N_SP, 512, N_SP);

  // 4. Cross-gate (F=512)
  crossgate_kernel<<<N_SP, 256, 0, stream>>>(H1, H2, g, 512);
  colnorm_kernel<float><<<4, 128, 0, stream>>>(H1, 512, g, H1, 512, N_SP, 512, 0.0f);
  colnorm_kernel<float><<<4, 128, 0, stream>>>(H2, 512, g, H2, 512, N_SP, 512, 0.0f);

  // 5. GCN layer 2 (512 -> 1024)
  colnorm_kernel<_Float16><<<4, 128, 0, stream>>>(H1, 512, nullptr, T1h, 512, N_SP, 512, BN_EPS);
  gemm_kernel<true, true,  false, true ><<<dim3(N_SP / 32, 1024 / 64), 256, 0, stream>>>(T1h, W11h, b1_1, T2h, N_SP, 1024, 512);
  gemm_kernel<true, false, true,  false><<<dim3(N_SP / 32, 1024 / 64), 256, 0, stream>>>(A1h, T2h, nullptr, H1, N_SP, 1024, N_SP);
  colnorm_kernel<_Float16><<<4, 128, 0, stream>>>(H2, 512, nullptr, T1h, 512, N_SP, 512, BN_EPS);
  gemm_kernel<true, true,  false, true ><<<dim3(N_SP / 32, 1024 / 64), 256, 0, stream>>>(T1h, W21h, b2_1, T2h, N_SP, 1024, 512);
  gemm_kernel<true, false, true,  false><<<dim3(N_SP / 32, 1024 / 64), 256, 0, stream>>>(A2h, T2h, nullptr, H2, N_SP, 1024, N_SP);

  // 6. Cross-gate (F=1024)
  crossgate_kernel<<<N_SP, 256, 0, stream>>>(H1, H2, g, 1024);
  colnorm_kernel<float><<<8, 128, 0, stream>>>(H1, 1024, g, H1, 1024, N_SP, 1024, 0.0f);
  colnorm_kernel<float><<<8, 128, 0, stream>>>(H2, 1024, g, H2, 1024, N_SP, 1024, 0.0f);

  // 7. Concat + BN (per-column, so BN(concat) == concat of per-half BN) -> f16
  colnorm_kernel<_Float16><<<8, 128, 0, stream>>>(H1, 1024, nullptr, Hch,        N_SP, N_SP, 1024, BN_EPS);
  colnorm_kernel<_Float16><<<8, 128, 0, stream>>>(H2, 1024, nullptr, Hch + 1024, N_SP, N_SP, 1024, BN_EPS);

  // 8. Output GCN layer (2048 -> 32), uses A1h
  gemm_kernel<true, true,  false, true><<<dim3(N_SP / 32, 1), 256, 0, stream>>>(Hch, Wouth, b_out, Touth, N_SP, 32, N_SP);
  gemm_kernel<true, false, true,  true><<<dim3(N_SP / 32, 1), 256, 0, stream>>>(A1h, Touth, nullptr, Houth, N_SP, 32, N_SP);

  // 9. Pixel projection: R = Q1 @ Hout (second 512 MB Q stream, f32 A path)
  gemm_kernel<false, false, false, false><<<dim3(HW_PIX / 32, 1), 256, 0, stream>>>(Q1, Houth, nullptr, R, HW_PIX, 32, N_SP);

  // 10. Classifier + softmax
  cls_softmax_kernel<<<HW_PIX / 256, 256, 0, stream>>>(R, W_cls, b_cls, Y);
}
